// ExemplarGNN2AdjModel_27152783245644
// MI455X (gfx1250) — matrix operand order repaired and checked by reading, not compiled, not run on hardware
//
#include <hip/hip_runtime.h>

typedef __attribute__((ext_vector_type(16))) _Float16 v16h;
typedef __attribute__((ext_vector_type(8)))  _Float16 v8h;
typedef __attribute__((ext_vector_type(8)))  float    v8f;

#define NB   512   // batch / num nodes
#define IND  595   // input dim
#define HID  128   // hidden dim

// ---------------------------------------------------------------------------
// Encoder layer 1: h1 = relu(x @ W1 + b1)     [512,595]x[595,128]
// ---------------------------------------------------------------------------
__global__ void __launch_bounds__(128) enc1_kernel(
    const float* __restrict__ x, const float* __restrict__ W1,
    const float* __restrict__ b1, float* __restrict__ h1) {
  __shared__ float xs[IND];
  const int b = blockIdx.x, n = threadIdx.x;
  for (int k = threadIdx.x; k < IND; k += 128) xs[k] = x[b * IND + k];
  __syncthreads();
  float acc = b1[n];
  for (int k = 0; k < IND; ++k) acc = fmaf(xs[k], W1[k * HID + n], acc);
  h1[b * HID + n] = fmaxf(acc, 0.0f);
}

// ---------------------------------------------------------------------------
// Encoder layer 2: h = relu(h1 @ W2 + b2); also emit h in f16 for WMMA
// ---------------------------------------------------------------------------
__global__ void __launch_bounds__(128) enc2_kernel(
    const float* __restrict__ h1, const float* __restrict__ W2,
    const float* __restrict__ b2, float* __restrict__ h,
    _Float16* __restrict__ hh) {
  __shared__ float hs[HID];
  const int b = blockIdx.x, n = threadIdx.x;
  hs[n] = h1[b * HID + n];
  __syncthreads();
  float acc = b2[n];
  for (int k = 0; k < HID; ++k) acc = fmaf(hs[k], W2[k * HID + n], acc);
  float r = fmaxf(acc, 0.0f);
  h[b * HID + n] = r;
  hh[b * HID + n] = (_Float16)r;
}

// ---------------------------------------------------------------------------
// Factored pair terms:
//   P[b,n] = bp1[n] + sum_k h[b,k] * Wp1[k      , n]   (hi block)
//   Q[b,n] =          sum_k h[b,k] * Wp1[HID + k, n]   (hj block)
// ---------------------------------------------------------------------------
__global__ void __launch_bounds__(128) prep_kernel(
    const float* __restrict__ h, const float* __restrict__ Wp1,
    const float* __restrict__ bp1, float* __restrict__ P,
    float* __restrict__ Q) {
  __shared__ float hs[HID];
  const int b = blockIdx.x, n = threadIdx.x;
  hs[n] = h[b * HID + n];
  __syncthreads();
  float p = bp1[n], q = 0.0f;
  for (int k = 0; k < HID; ++k) {
    p = fmaf(hs[k], Wp1[k * HID + n], p);
    q = fmaf(hs[k], Wp1[(HID + k) * HID + n], q);
  }
  P[b * HID + n] = p;
  Q[b * HID + n] = q;
}

// ---------------------------------------------------------------------------
// Transpose + f16-convert the |hi-hj| weight block: WcT[n][k] = Wp1[2H+k][n]
// ---------------------------------------------------------------------------
__global__ void __launch_bounds__(128) convw_kernel(
    const float* __restrict__ Wp1, _Float16* __restrict__ WcT) {
  const int n = blockIdx.x, k = threadIdx.x;
  WcT[n * HID + k] = (_Float16)Wp1[(2 * HID + k) * HID + n];
}

// ---------------------------------------------------------------------------
// Main pair kernel. grid = (4 j-blocks, 512 i), block = 128 (4 wave32).
// Per WG: C = |h_i - h_jblk| @ Wc  (128x128x128 GEMM via v_wmma_f32_16x16x32_f16)
// then scores[i, jblk] = relu(C + P[i] + Q[j]) . Wp2 + bp2, fused.
//
// Fragments gathered straight from global (hot in WGP$/L2 across all 2048 WGs);
// no LDS staging -> no bank-conflicted ds gathers, LDS = 512B (h_i broadcast).
// All 8 B fragments of a K-step preloaded, so 16 WMMAs issue back-to-back.
// ---------------------------------------------------------------------------
__global__ void __launch_bounds__(128) pair_kernel(
    const _Float16* __restrict__ hh, const _Float16* __restrict__ WcT,
    const float* __restrict__ P, const float* __restrict__ Q,
    const float* __restrict__ Wp2, const float* __restrict__ bp2,
    float* __restrict__ out) {
  __shared__ float hi_s[HID];   // h_i row as f32, broadcast reads only

  const int i     = blockIdx.y;
  const int jbase = blockIdx.x * 128;
  const int tid   = threadIdx.x;

  hi_s[tid] = (float)hh[(size_t)i * HID + tid];
  __syncthreads();

  const int wave = tid >> 5;
  const int lane = tid & 31;
  const int lh   = lane >> 4;   // lane half (0: lanes 0-15, 1: lanes 16-31)
  const int ln   = lane & 15;

  // per-lane source rows (256B-aligned, 16B-aligned sub-loads)
  const _Float16* hrow0 = hh + (size_t)(jbase + (wave * 2 + 0) * 16 + ln) * HID;
  const _Float16* hrow1 = hh + (size_t)(jbase + (wave * 2 + 1) * 16 + ln) * HID;

  v8f acc[2][8];
#pragma unroll
  for (int m = 0; m < 2; ++m)
#pragma unroll
    for (int nt = 0; nt < 8; ++nt) acc[m][nt] = {};

#pragma unroll
  for (int kk = 0; kk < 4; ++kk) {        // K = 128 in 4 steps of 32
    const int kb = kk * 32;

    // ---- preload all 8 B fragments for this K-step (global_load_b128 x16) ----
    // B 32x16 fragment, contiguous K order used here: N = lane%16, element e <-> K = e + lh*16
    v16h bfrag[8];
#pragma unroll
    for (int nt = 0; nt < 8; ++nt) {
      const _Float16* wrow = WcT + (size_t)(nt * 16 + ln) * HID + kb + lh * 16;
      v8h blo = *(const v8h*)(wrow);
      v8h bhi = *(const v8h*)(wrow + 8);
#pragma unroll
      for (int e = 0; e < 8; ++e) { bfrag[nt][e] = blo[e]; bfrag[nt][8 + e] = bhi[e]; }
    }

    // ---- build 2 A fragments: 16x32, contiguous K order (same as B) ----
    // lanes 0-15: K {0-15}; lanes 16-31: K {16-31}
    v16h afrag[2];
#pragma unroll
    for (int m = 0; m < 2; ++m) {
      const _Float16* hrow = (m == 0) ? hrow0 : hrow1;
      const int kA = kb + lh * 16;
      v8h a0 = *(const v8h*)(hrow + kA);        // K = kA .. kA+7
      v8h a1 = *(const v8h*)(hrow + kA + 8);    // K = kA+8 .. kA+15
#pragma unroll
      for (int e = 0; e < 8; ++e) {
        afrag[m][e]     = (_Float16)fabsf(hi_s[kA + e]      - (float)a0[e]);
        afrag[m][8 + e] = (_Float16)fabsf(hi_s[kA + 8 + e] - (float)a1[e]);
      }
    }

    // ---- 16 WMMAs back-to-back ----
#pragma unroll
    for (int nt = 0; nt < 8; ++nt) {
      acc[0][nt] = __builtin_amdgcn_wmma_f32_16x16x32_f16(
          false, afrag[0], false, bfrag[nt], (short)0, acc[0][nt], false, false);
      acc[1][nt] = __builtin_amdgcn_wmma_f32_16x16x32_f16(
          false, afrag[1], false, bfrag[nt], (short)0, acc[1][nt], false, false);
    }
  }

  // Fused epilogue: relu(C + P[i] + Q[j]) . Wp2 + bp2, reduce over n.
  float Pr[8], Wr[8];
#pragma unroll
  for (int nt = 0; nt < 8; ++nt) {
    const int n = nt * 16 + ln;
    Pr[nt] = P[i * HID + n];
    Wr[nt] = Wp2[n];
  }
  const float bias2 = bp2[0];

#pragma unroll
  for (int m = 0; m < 2; ++m) {
    const int mtile = wave * 2 + m;
    float s[8];
#pragma unroll
    for (int r = 0; r < 8; ++r) {     // C layout: M = r + lh*8, N = ln + nt*16
      const int j = jbase + mtile * 16 + lh * 8 + r;
      float sum = 0.0f;
#pragma unroll
      for (int nt = 0; nt < 8; ++nt) {
        const int n = nt * 16 + ln;
        float v = acc[m][nt][r] + Pr[nt] + Q[j * HID + n];
        sum = fmaf(fmaxf(v, 0.0f), Wr[nt], sum);
      }
      s[r] = sum;
    }
#pragma unroll
    for (int r = 0; r < 8; ++r) {     // reduce across the 16 lanes of each half
      float v = s[r];
      v += __shfl_xor(v, 1, 32);
      v += __shfl_xor(v, 2, 32);
      v += __shfl_xor(v, 4, 32);
      v += __shfl_xor(v, 8, 32);
      s[r] = v;
    }
    if (ln == 0) {
#pragma unroll
      for (int r = 0; r < 8; ++r) {
        const int j = jbase + mtile * 16 + lh * 8 + r;
        out[(size_t)i * NB + j] = s[r] + bias2;
      }
    }
  }
}

// ---------------------------------------------------------------------------
extern "C" void kernel_launch(void* const* d_in, const int* in_sizes, int n_in,
                              void* d_out, int out_size, void* d_ws, size_t ws_size,
                              hipStream_t stream) {
  const float* x   = (const float*)d_in[0];
  const float* W1  = (const float*)d_in[1];
  const float* b1  = (const float*)d_in[2];
  const float* W2  = (const float*)d_in[3];
  const float* b2  = (const float*)d_in[4];
  const float* Wp1 = (const float*)d_in[5];
  const float* bp1 = (const float*)d_in[6];
  const float* Wp2 = (const float*)d_in[7];
  const float* bp2 = (const float*)d_in[8];
  float* out = (float*)d_out;

  char* ws = (char*)d_ws;
  size_t off = 0;
  float*    h1  = (float*)(ws + off);    off += (size_t)NB * HID * sizeof(float);
  float*    h   = (float*)(ws + off);    off += (size_t)NB * HID * sizeof(float);
  float*    P   = (float*)(ws + off);    off += (size_t)NB * HID * sizeof(float);
  float*    Q   = (float*)(ws + off);    off += (size_t)NB * HID * sizeof(float);
  _Float16* hh  = (_Float16*)(ws + off); off += (size_t)NB * HID * sizeof(_Float16);
  _Float16* WcT = (_Float16*)(ws + off); off += (size_t)HID * HID * sizeof(_Float16);
  (void)off; (void)ws_size; (void)in_sizes; (void)n_in; (void)out_size;

  enc1_kernel<<<NB, 128, 0, stream>>>(x, W1, b1, h1);
  enc2_kernel<<<NB, 128, 0, stream>>>(h1, W2, b2, h, hh);
  prep_kernel<<<NB, 128, 0, stream>>>(h, Wp1, bp1, P, Q);
  convw_kernel<<<HID, 128, 0, stream>>>(Wp1, WcT);
  pair_kernel<<<dim3(NB / 128, NB), 128, 0, stream>>>(hh, WcT, P, Q, Wp2, bp2, out);
}
